// Transformer_7413113553610
// MI455X (gfx1250) — matrix operand order repaired and checked by reading, not compiled, not run on hardware
//
#include <hip/hip_runtime.h>
#include <hip/hip_bf16.h>

typedef __attribute__((ext_vector_type(16))) _Float16 v16h;
typedef __attribute__((ext_vector_type(8)))  _Float16 v8h;
typedef __attribute__((ext_vector_type(8)))  float    v8f;
typedef __attribute__((ext_vector_type(4)))  float    v4f;
typedef __attribute__((ext_vector_type(4)))  unsigned int u32x4;
typedef __attribute__((ext_vector_type(8)))  int      i32x8;
typedef __attribute__((ext_vector_type(4)))  int      i32x4;

#define D_MODEL 768
#define NROWS   2048   // B*T
#define TSEQ    512
#define NHEAD   12
#define DHEAD   64
#define NLAYER  12
#define VOCAB   32000
#define LN_EPS  1e-5f

#if __has_builtin(__builtin_amdgcn_tensor_load_to_lds) && \
    __has_builtin(__builtin_amdgcn_s_wait_tensorcnt)
#define HAS_TDM 1
#else
#define HAS_TDM 0
#endif

static __device__ __forceinline__ v8f wmma16(v16h a, v16h b, v8f c) {
  // D = A(16x32 f16) * B(32x16 f16) + C(16x16 f32)
  return __builtin_amdgcn_wmma_f32_16x16x32_f16(false, a, false, b, (short)0, c,
                                                false, false);
}
static __device__ __forceinline__ v16h join16(v8h lo, v8h hi) {
  v16h r;
#pragma unroll
  for (int i = 0; i < 8; ++i) { r[i] = lo[i]; r[i + 8] = hi[i]; }
  return r;
}
static __device__ __forceinline__ v8f zero8() {
  v8f z;
#pragma unroll
  for (int i = 0; i < 8; ++i) z[i] = 0.0f;
  return z;
}

#if HAS_TDM
// ---------------------------------------------------------------------------
// Tensor Data Mover: async copy of a 2-D f16 tile (tile_w x tile_h elements,
// global row stride row_stride elements) into LDS, padding each row from
// tile_w halves to tile_w+8 halves (pad_amount = 4 DWORDs).
// D# layout per CDNA5 ISA ch.8 (group0 128b, group1 256b, groups 2/3 zero).
// ---------------------------------------------------------------------------
static __device__ __forceinline__ void tdm_load_2d_f16(
    unsigned lds_off, const void* gaddr, unsigned tile_w, unsigned tile_h,
    unsigned row_stride, unsigned pad_interval_code, unsigned pad_amount_code) {
  unsigned long long ga = (unsigned long long)(size_t)gaddr;
  u32x4 g0;
  g0[0] = 1u;                                       // count=1 (valid), user
  g0[1] = lds_off;                                  // lds_addr (bytes)
  g0[2] = (unsigned)(ga & 0xFFFFFFFFu);             // global_addr[31:0]
  g0[3] = (unsigned)((ga >> 32) & 0x01FFFFFFu)      // global_addr[56:32]
          | (2u << 30);                             // type = 2 ("image")
  i32x8 g1;
  g1[0] = (int)((1u << 16)                          // data_size = 1 (2 bytes)
                | (1u << 20)                        // pad_enable
                | (pad_interval_code << 22)         // DWORDs between pads
                | (pad_amount_code << 25));         // pad DWORDs - 1
  g1[1] = (int)((tile_w & 0xFFFFu) << 16);          // tensor_dim0[15:0]
  g1[2] = (int)(((tile_w >> 16) & 0xFFFFu)          // tensor_dim0[31:16]
                | ((tile_h & 0xFFFFu) << 16));      // tensor_dim1[15:0]
  g1[3] = (int)(((tile_h >> 16) & 0xFFFFu)          // tensor_dim1[31:16]
                | ((tile_w & 0xFFFFu) << 16));      // tile_dim0
  g1[4] = (int)(tile_h & 0xFFFFu);                  // tile_dim1 (tile_dim2=0)
  g1[5] = (int)row_stride;                          // tensor_dim0_stride[31:0]
  g1[6] = 0;                                        // stride[47:32], d1s[15:0]
  g1[7] = 0;
  i32x4 gz;
  gz[0] = 0; gz[1] = 0; gz[2] = 0; gz[3] = 0;
#if __clang_major__ >= 23
  i32x8 gz8;
#pragma unroll
  for (int i = 0; i < 8; ++i) gz8[i] = 0;
  __builtin_amdgcn_tensor_load_to_lds(g0, g1, gz, gz, gz8, 0);
#else
  __builtin_amdgcn_tensor_load_to_lds(g0, g1, gz, gz, 0);
#endif
}
#endif  // HAS_TDM

// ---------------------------------------------------------------------------
// WMMA GEMM:  C[M,N] = act( A_f16[M,K] @ B_f32[K,N] + bias )
// BM=128 BN=128 BK=32, 256 threads (8 waves); wave -> 32x64 strip (8 WMMAs).
// Software-pipelined: global loads for tile k+1 issued before computing tile k.
// QKV=1: B gathered from three [H,D,Dh] tensors (fused QKV, N = 3*768).
// ---------------------------------------------------------------------------
template <int QKV>
__global__ __launch_bounds__(256) void gemm_wmma_kernel(
    const _Float16* __restrict__ A, int lda,
    const float* __restrict__ B0, const float* __restrict__ B1,
    const float* __restrict__ B2, int ldb, int N, int K,
    const float* __restrict__ bias, int relu,
    float* __restrict__ out32, _Float16* __restrict__ out16) {
  __shared__ __align__(16) _Float16 As[128][40];   // row-major, pad to 40
  __shared__ __align__(16) _Float16 Bs[128][40];   // TRANSPOSED: Bs[n][k]

  const int tid  = threadIdx.x;
  const int lane = tid & 31;
  const int wave = tid >> 5;
  const int m_base = blockIdx.y * 128;
  const int n_base = blockIdx.x * 128;
  const int m0 = (wave & 3) * 32;
  const int n0 = (wave >> 2) * 64;
  const int lm = lane & 15;
  const int lh = lane >> 4;

  // ---- staging index precompute (hoisted out of k-loop) ----
  int arow[2], acol[2];
  const _Float16* aptr[2];
#pragma unroll
  for (int i = 0; i < 2; ++i) {
    int g = tid + 256 * i;          // 0..511
    arow[i] = g >> 2;               // 0..127
    acol[i] = (g & 3) * 8;          // 0,8,16,24
    aptr[i] = A + (size_t)(m_base + arow[i]) * lda + acol[i];
  }
  int bk[4], bn4[4];
  const float* bptr[4];
  int bstride;
#pragma unroll
  for (int i = 0; i < 4; ++i) {
    int g = tid + 256 * i;          // 0..1023
    bk[i]  = g >> 5;                // 0..31
    bn4[i] = (g & 31) * 4;          // 0..124
    int gn = n_base + bn4[i];
    if (QKV) {
      int sel = gn / 768;
      int c   = gn - sel * 768;
      int hh  = c >> 6;
      int ee  = c & 63;
      const float* W = (sel == 0) ? B0 : (sel == 1) ? B1 : B2;
      bptr[i] = W + (size_t)hh * (768 * 64) + ee + (size_t)bk[i] * 64;
    } else {
      bptr[i] = B0 + (size_t)bk[i] * ldb + gn;
    }
  }
  bstride = QKV ? 64 : ldb;

  v8f acc[2][4];
#pragma unroll
  for (int h2 = 0; h2 < 2; ++h2)
#pragma unroll
    for (int j = 0; j < 4; ++j) acc[h2][j] = zero8();

  v8h areg[2];
  v4f breg[4];

  // prologue: stage tile k0=0 into registers
#pragma unroll
  for (int i = 0; i < 2; ++i) areg[i] = *(const v8h*)(aptr[i]);
#pragma unroll
  for (int i = 0; i < 4; ++i) breg[i] = *(const v4f*)(bptr[i]);

  for (int k0 = 0; k0 < K; k0 += 32) {
    // commit staged registers to LDS
#pragma unroll
    for (int i = 0; i < 2; ++i) *(v8h*)&As[arow[i]][acol[i]] = areg[i];
#pragma unroll
    for (int i = 0; i < 4; ++i) {
#pragma unroll
      for (int j = 0; j < 4; ++j) Bs[bn4[i] + j][bk[i]] = (_Float16)breg[i][j];
    }
    __syncthreads();

    // issue next tile's global loads (overlap with WMMA below)
    if (k0 + 32 < K) {
#pragma unroll
      for (int i = 0; i < 2; ++i)
        areg[i] = *(const v8h*)(aptr[i] + (k0 + 32));
#pragma unroll
      for (int i = 0; i < 4; ++i)
        breg[i] = *(const v4f*)(bptr[i] + (size_t)(k0 + 32) * bstride);
    }

    // compute: 2 A-frags x 4 B-frags = 8 WMMAs
    v16h af[2];
#pragma unroll
    for (int h2 = 0; h2 < 2; ++h2) {
      const _Float16* ar = &As[m0 + 16 * h2 + lm][0];
      af[h2] = join16(*(const v8h*)(ar + 8 * lh),
                      *(const v8h*)(ar + 16 + 8 * lh));
    }
#pragma unroll
    for (int j = 0; j < 4; ++j) {
      const _Float16* br = &Bs[n0 + 16 * j + lm][0];
      v16h bf = join16(*(const v8h*)(br + 8 * lh),
                       *(const v8h*)(br + 16 + 8 * lh));
      acc[0][j] = wmma16(af[0], bf, acc[0][j]);
      acc[1][j] = wmma16(af[1], bf, acc[1][j]);
    }
    __syncthreads();
  }

  // epilogue: bias / relu / store f32 and/or f16
#pragma unroll
  for (int j = 0; j < 4; ++j) {
    int c = n_base + n0 + 16 * j + lm;
    float bb = bias ? bias[c] : 0.0f;
#pragma unroll
    for (int h2 = 0; h2 < 2; ++h2) {
#pragma unroll
      for (int r = 0; r < 8; ++r) {
        int row = m_base + m0 + 16 * h2 + r + 8 * lh;
        float v = acc[h2][j][r] + bb;
        if (relu) v = fmaxf(v, 0.0f);
        if (out32) out32[(size_t)row * N + c] = v;
        if (out16) out16[(size_t)row * N + c] = (_Float16)v;
      }
    }
  }
}

// ---------------------------------------------------------------------------
// Flash attention: block = (qt, h, b), 128 threads (4 waves), 64 query rows,
// online softmax over causal key tiles of 64.  qkv layout [N, 3*768] f16.
// Q and K tiles arrive via the Tensor Data Mover (async, TENSORcnt) when
// available; V is loaded manually (needs a transpose) concurrently.
// ---------------------------------------------------------------------------
__global__ __launch_bounds__(128) void attn_kernel(
    const _Float16* __restrict__ qkv, _Float16* __restrict__ attn_out) {
  __shared__ __align__(16) _Float16 Qs[64][72];
  __shared__ __align__(16) _Float16 Ks[64][72];       // [key][dh]
  __shared__ __align__(16) _Float16 Vts[64][72];      // transposed [dh][key]
  __shared__ __align__(16) _Float16 Ps[4][16][72];    // per-wave P tile

  const int tid  = threadIdx.x;
  const int lane = tid & 31;
  const int wave = tid >> 5;   // 0..3
  const int lm = lane & 15, lh = lane >> 4;
  const int qt = blockIdx.x;   // query tile 0..7
  const int h  = blockIdx.y;
  const int b  = blockIdx.z;

  const size_t rs = 3 * D_MODEL;  // 2304 halves per token row
  const _Float16* Qg = qkv + ((size_t)b * TSEQ) * rs + h * DHEAD;
  const _Float16* Kg = Qg + D_MODEL;
  const _Float16* Vg = Qg + 2 * D_MODEL;

  // ---- load Q tile ----
#if HAS_TDM
  if (wave == 0) {
    tdm_load_2d_f16((unsigned)(size_t)&Qs[0][0],
                    Qg + (size_t)(qt * 64) * rs, 64, 64, (unsigned)rs,
                    /*interval 32 DWORDs*/ 4u, /*pad 4 DWORDs*/ 3u);
    __builtin_amdgcn_s_wait_tensorcnt(0);
  }
#else
#pragma unroll
  for (int i = 0; i < 4; ++i) {
    int g = tid + 128 * i;             // 0..511
    int r = g >> 3, c8 = (g & 7) * 8;  // row 0..63, col 0..56
    *(v8h*)&Qs[r][c8] = *(const v8h*)(Qg + (size_t)(qt * 64 + r) * rs + c8);
  }
#endif
  __syncthreads();

  v16h aq[2];
#pragma unroll
  for (int kk = 0; kk < 2; ++kk) {
    const _Float16* qr = &Qs[wave * 16 + lm][0];
    aq[kk] = join16(*(const v8h*)(qr + 32 * kk + 8 * lh),
                    *(const v8h*)(qr + 32 * kk + 16 + 8 * lh));
  }

  v8f accO[4];
#pragma unroll
  for (int j = 0; j < 4; ++j) accO[j] = zero8();
  float mrow[8], srow[8];
#pragma unroll
  for (int r = 0; r < 8; ++r) { mrow[r] = -1e30f; srow[r] = 0.0f; }

  const float scale = 0.03608439182435161f;  // 1/sqrt(768) (reference quirk)

  for (int kt = 0; kt <= qt; ++kt) {
    // ---- stage K tile (TDM) + V tile (manual transpose), overlapped ----
#if HAS_TDM
    if (wave == 0)
      tdm_load_2d_f16((unsigned)(size_t)&Ks[0][0],
                      Kg + (size_t)(kt * 64) * rs, 64, 64, (unsigned)rs,
                      4u, 3u);
#endif
#pragma unroll
    for (int i = 0; i < 4; ++i) {
      int g = tid + 128 * i;
      int r = g >> 3, c8 = (g & 7) * 8;
#if !HAS_TDM
      *(v8h*)&Ks[r][c8] = *(const v8h*)(Kg + (size_t)(kt * 64 + r) * rs + c8);
#endif
      v8h vv = *(const v8h*)(Vg + (size_t)(kt * 64 + r) * rs + c8);
#pragma unroll
      for (int e = 0; e < 8; ++e) Vts[c8 + e][r] = vv[e];
    }
#if HAS_TDM
    if (wave == 0) __builtin_amdgcn_s_wait_tensorcnt(0);
#endif
    __syncthreads();

    // S = Q @ K^T  (16 rows x 64 keys per wave)
    v8f s[4];
#pragma unroll
    for (int j = 0; j < 4; ++j) s[j] = zero8();
#pragma unroll
    for (int kk = 0; kk < 2; ++kk)
#pragma unroll
      for (int j = 0; j < 4; ++j) {
        const _Float16* kr = &Ks[16 * j + lm][0];
        v16h bk = join16(*(const v8h*)(kr + 32 * kk + 8 * lh),
                         *(const v8h*)(kr + 32 * kk + 16 + 8 * lh));
        s[j] = wmma16(aq[kk], bk, s[j]);
      }

    // scale + causal mask + row max
    float rmax[8];
#pragma unroll
    for (int r = 0; r < 8; ++r) rmax[r] = -1e30f;
#pragma unroll
    for (int j = 0; j < 4; ++j) {
      int key = kt * 64 + 16 * j + lm;
#pragma unroll
      for (int r = 0; r < 8; ++r) {
        int qrow = qt * 64 + wave * 16 + r + 8 * lh;
        float v = s[j][r] * scale;
        if (key > qrow) v = -1e30f;
        s[j][r] = v;
        rmax[r] = fmaxf(rmax[r], v);
      }
    }
#pragma unroll
    for (int r = 0; r < 8; ++r)
#pragma unroll
      for (int msk = 1; msk <= 8; msk <<= 1)
        rmax[r] = fmaxf(rmax[r], __shfl_xor(rmax[r], msk, 32));

    // online softmax rescale
    float padd[8];
#pragma unroll
    for (int r = 0; r < 8; ++r) {
      float mnew = fmaxf(mrow[r], rmax[r]);
      float cf = __expf(mrow[r] - mnew);
      mrow[r] = mnew;
      srow[r] *= cf;
#pragma unroll
      for (int j = 0; j < 4; ++j) accO[j][r] *= cf;
      padd[r] = 0.0f;
    }
#pragma unroll
    for (int j = 0; j < 4; ++j)
#pragma unroll
      for (int r = 0; r < 8; ++r) {
        float p = __expf(s[j][r] - mrow[r]);
        padd[r] += p;
        Ps[wave][r + 8 * lh][16 * j + lm] = (_Float16)p;
      }
#pragma unroll
    for (int r = 0; r < 8; ++r) {
#pragma unroll
      for (int msk = 1; msk <= 8; msk <<= 1)
        padd[r] += __shfl_xor(padd[r], msk, 32);
      srow[r] += padd[r];
    }
    __syncthreads();

    // O += P @ V
#pragma unroll
    for (int kk = 0; kk < 2; ++kk) {
      const _Float16* pr = &Ps[wave][lm][0];
      v16h ap = join16(*(const v8h*)(pr + 32 * kk + 8 * lh),
                       *(const v8h*)(pr + 32 * kk + 16 + 8 * lh));
#pragma unroll
      for (int j = 0; j < 4; ++j) {
        const _Float16* vr = &Vts[16 * j + lm][0];
        v16h bv = join16(*(const v8h*)(vr + 32 * kk + 8 * lh),
                         *(const v8h*)(vr + 32 * kk + 16 + 8 * lh));
        accO[j] = wmma16(ap, bv, accO[j]);
      }
    }
    __syncthreads();
  }

  // normalize + store concat-heads layout [N, 768]
#pragma unroll
  for (int r = 0; r < 8; ++r) {
    float inv = srow[r] > 0.0f ? 1.0f / srow[r] : 0.0f;
    size_t row = (size_t)b * TSEQ + qt * 64 + wave * 16 + r + 8 * lh;
#pragma unroll
    for (int j = 0; j < 4; ++j) {
      int col = h * DHEAD + 16 * j + lm;
      attn_out[row * D_MODEL + col] = (_Float16)(accO[j][r] * inv);
    }
  }
}

// ---------------------------------------------------------------------------
// x = embed[tok] + sinusoidal PE ; writes f32 master + f16 copy
// ---------------------------------------------------------------------------
__global__ __launch_bounds__(256) void embed_kernel(
    const int* __restrict__ tokens, const float* __restrict__ embed,
    float* __restrict__ x32, _Float16* __restrict__ x16) {
  int row = blockIdx.x;
  int t = row & (TSEQ - 1);
  int tok = tokens[row];
#pragma unroll
  for (int i = 0; i < 3; ++i) {
    int d = threadIdx.x + 256 * i;
    float ex = embed[(size_t)tok * D_MODEL + d];
    int i2 = d >> 1;
    float div = __powf(10000.0f, (float)(2 * i2) * (1.0f / D_MODEL));
    float ang = (float)t / div;
    float pe = (d & 1) ? __cosf(ang) : __sinf(ang);
    float v = ex + pe;
    x32[(size_t)row * D_MODEL + d] = v;
    x16[(size_t)row * D_MODEL + d] = (_Float16)v;
  }
}

// ---------------------------------------------------------------------------
// x += LayerNorm(h) * gamma + beta ; refresh f16 copy
// ---------------------------------------------------------------------------
__global__ __launch_bounds__(256) void ln_residual_kernel(
    const float* __restrict__ h, const float* __restrict__ gamma,
    const float* __restrict__ beta, float* __restrict__ x32,
    _Float16* __restrict__ x16) {
  __shared__ float rsum[256];
  __shared__ float rsq[256];
  int row = blockIdx.x;
  float v[3];
  float s = 0.0f, sq = 0.0f;
#pragma unroll
  for (int i = 0; i < 3; ++i) {
    v[i] = h[(size_t)row * D_MODEL + threadIdx.x + 256 * i];
    s += v[i];
    sq += v[i] * v[i];
  }
  rsum[threadIdx.x] = s;
  rsq[threadIdx.x] = sq;
  __syncthreads();
  for (int off = 128; off > 0; off >>= 1) {
    if (threadIdx.x < off) {
      rsum[threadIdx.x] += rsum[threadIdx.x + off];
      rsq[threadIdx.x] += rsq[threadIdx.x + off];
    }
    __syncthreads();
  }
  float mean = rsum[0] * (1.0f / D_MODEL);
  float var = rsq[0] * (1.0f / D_MODEL) - mean * mean;
  float rstd = rsqrtf(var + LN_EPS);
#pragma unroll
  for (int i = 0; i < 3; ++i) {
    int d = threadIdx.x + 256 * i;
    float y = (v[i] - mean) * rstd * gamma[d] + beta[d];
    float nx = x32[(size_t)row * D_MODEL + d] + y;
    x32[(size_t)row * D_MODEL + d] = nx;
    x16[(size_t)row * D_MODEL + d] = (_Float16)nx;
  }
}

// ---------------------------------------------------------------------------
extern "C" void kernel_launch(void* const* d_in, const int* in_sizes, int n_in,
                              void* d_out, int out_size, void* d_ws,
                              size_t ws_size, hipStream_t stream) {
  (void)in_sizes; (void)n_in; (void)out_size; (void)ws_size;
  const int*   tokens = (const int*)d_in[0];
  const float* embedW = (const float*)d_in[1];
  const float* Wq     = (const float*)d_in[2];
  const float* Wk     = (const float*)d_in[3];
  const float* Wv     = (const float*)d_in[4];
  const float* Wo     = (const float*)d_in[5];
  const float* bo     = (const float*)d_in[6];
  const float* ln1g   = (const float*)d_in[7];
  const float* ln1b   = (const float*)d_in[8];
  const float* W1     = (const float*)d_in[9];
  const float* bf1    = (const float*)d_in[10];
  const float* W2     = (const float*)d_in[11];
  const float* bf2    = (const float*)d_in[12];
  const float* ln2g   = (const float*)d_in[13];
  const float* ln2b   = (const float*)d_in[14];
  const float* Wout   = (const float*)d_in[15];
  const float* bout   = (const float*)d_in[16];

  char* w = (char*)d_ws;
  float*    x32    = (float*)w;    w += (size_t)NROWS * D_MODEL * 4;
  _Float16* x16    = (_Float16*)w; w += (size_t)NROWS * D_MODEL * 2;
  _Float16* qkv16  = (_Float16*)w; w += (size_t)NROWS * 3 * D_MODEL * 2;
  _Float16* attn16 = (_Float16*)w; w += (size_t)NROWS * D_MODEL * 2;
  float*    sub32  = (float*)w;    w += (size_t)NROWS * D_MODEL * 4;
  _Float16* h16    = (_Float16*)w; w += (size_t)NROWS * D_MODEL * 2;

  embed_kernel<<<NROWS, 256, 0, stream>>>(tokens, embedW, x32, x16);

  const size_t WSL = (size_t)D_MODEL * D_MODEL;  // per-layer weight slice
  const dim3 gQKV(3 * D_MODEL / 128, NROWS / 128);
  const dim3 gSQ(D_MODEL / 128, NROWS / 128);
  const dim3 gOUT(VOCAB / 128, NROWS / 128);

  for (int l = 0; l < NLAYER; ++l) {
    // fused QKV projection: [2048,768] x [768,2304] -> qkv16
    gemm_wmma_kernel<1><<<gQKV, 256, 0, stream>>>(
        x16, D_MODEL, Wq + l * WSL, Wk + l * WSL, Wv + l * WSL, 0,
        3 * D_MODEL, D_MODEL, nullptr, 0, nullptr, qkv16);
    // causal flash attention -> attn16 [2048,768]
    attn_kernel<<<dim3(TSEQ / 64, NHEAD, 4), 128, 0, stream>>>(qkv16, attn16);
    // output projection + bias -> sub32
    gemm_wmma_kernel<0><<<gSQ, 256, 0, stream>>>(
        attn16, D_MODEL, Wo + l * WSL, nullptr, nullptr, D_MODEL, D_MODEL,
        D_MODEL, bo + l * D_MODEL, 0, sub32, nullptr);
    ln_residual_kernel<<<NROWS, 256, 0, stream>>>(
        sub32, ln1g + l * D_MODEL, ln1b + l * D_MODEL, x32, x16);
    // MLP
    gemm_wmma_kernel<0><<<gSQ, 256, 0, stream>>>(
        x16, D_MODEL, W1 + l * WSL, nullptr, nullptr, D_MODEL, D_MODEL,
        D_MODEL, bf1 + l * D_MODEL, 1, nullptr, h16);
    gemm_wmma_kernel<0><<<gSQ, 256, 0, stream>>>(
        h16, D_MODEL, W2 + l * WSL, nullptr, nullptr, D_MODEL, D_MODEL,
        D_MODEL, bf2 + l * D_MODEL, 0, sub32, nullptr);
    ln_residual_kernel<<<NROWS, 256, 0, stream>>>(
        sub32, ln2g + l * D_MODEL, ln2b + l * D_MODEL, x32, x16);
  }

  // logits = x @ Wout + bout  -> d_out [2048, 32000] f32
  gemm_wmma_kernel<0><<<gOUT, 256, 0, stream>>>(
      x16, D_MODEL, Wout, nullptr, nullptr, VOCAB, VOCAB, D_MODEL, bout, 0,
      (float*)d_out, nullptr);
}
